// GATr_model_34308198761147
// MI455X (gfx1250) — matrix operand order repaired and checked by reading, not compiled, not run on hardware
//
#include <hip/hip_runtime.h>

// GATr forward for MI455X (gfx1250, wave32).
// Compute-dominant term: 4096x4096 attention (~64 GFLOP) -> flash attention
// with v_wmma_f32_16x16x32_f16. All tensors << 192MB L2. To avoid being
// L2-bandwidth-bound, each wave now processes 32 query rows (two A tiles),
// reusing every K/V B-fragment for two WMMAs: 22 WMMA per 22 b128 loads/iter.

typedef _Float16 half8 __attribute__((ext_vector_type(8)));
typedef _Float16 v16h  __attribute__((ext_vector_type(16)));
typedef float    v8f   __attribute__((ext_vector_type(8)));

#define B_    4
#define N_    4096
#define H_    5
#define TOK   (B_ * N_)
#define DPAD  96      // 80 features padded to 3x32 for WMMA k-steps
#define DFEAT 80

// Blade bookkeeping (masks sorted by (popcount, value), metric (0,1,1,1))
__constant__ int   c_MASK[16]  = {0,1,2,4,8,3,5,6,9,10,12,7,11,13,14,15};
__constant__ int   c_IDX[16]   = {0,1,2,5,3,6,7,11,4,8,9,12,10,13,14,15};
__constant__ int   c_GRADE[16] = {0,1,1,1,1,2,2,2,2,2,2,3,3,3,3,4};
__constant__ float c_INNER[16] = {1,0,1,1,1,0,0,1,0,1,1,0,0,0,1,0};

// ---------------- embed: x(35) -> mv(32x16) -> w_in -> h(5x16) --------------
__global__ void embed_kernel(const float* __restrict__ x,
                             const float* __restrict__ w_in,
                             float* __restrict__ h) {
  int tid = blockIdx.x * blockDim.x + threadIdx.x;
  int t = tid >> 4, c = tid & 15;
  if (t >= TOK) return;
  const float* xr = x + (size_t)t * 35;
  const float* W = w_in + c_GRADE[c] * H_ * 32;  // (o,i)
  float y[H_] = {0.f, 0.f, 0.f, 0.f, 0.f};
#pragma unroll
  for (int i = 0; i < 32; ++i) {
    float mv;
    if (i == 0) {
      mv = (c == 0) ? 1.0f :
           (c == 5) ? xr[32] :
           (c == 6) ? xr[33] :
           (c == 8) ? xr[34] : 0.f;
    } else {
      mv = (c == 0) ? xr[i] : 0.f;
    }
#pragma unroll
    for (int o = 0; o < H_; ++o) y[o] += mv * W[o * 32 + i];
  }
#pragma unroll
  for (int o = 0; o < H_; ++o) h[(size_t)t * DFEAT + o * 16 + c] = y[o];
}

// ------------- layernorm + q/k/v projection (q gets INNER*scale) ------------
__global__ void ln_qkv_kernel(const float* __restrict__ h,
                              const float* __restrict__ wq,
                              const float* __restrict__ wk,
                              const float* __restrict__ wv,
                              _Float16* __restrict__ qb,
                              _Float16* __restrict__ kb,
                              _Float16* __restrict__ vT) {
  int tid = blockIdx.x * blockDim.x + threadIdx.x;
  int t = tid >> 4, c = tid & 15;
  if (t >= TOK) return;
  const float* hr = h + (size_t)t * DFEAT;
  float inner = c_INNER[c];
  float xc[H_];
  float ip = 0.f;
#pragma unroll
  for (int i = 0; i < H_; ++i) { xc[i] = hr[i * 16 + c]; ip += xc[i] * xc[i] * inner; }
#pragma unroll
  for (int m = 1; m < 16; m <<= 1) ip += __shfl_xor(ip, m, 32);
  float inv = rsqrtf(ip * (1.0f / (H_ * 16.0f)) + 1e-6f);
#pragma unroll
  for (int i = 0; i < H_; ++i) xc[i] *= inv;

  int g = c_GRADE[c];
  const float* Wq = wq + g * H_ * H_;
  const float* Wk = wk + g * H_ * H_;
  const float* Wv = wv + g * H_ * H_;
  float qscale = rsqrtf(16.0f * (float)H_) * inner;  // fold INNER + 1/sqrt(80)
  int b = t / N_, n = t % N_;
#pragma unroll
  for (int o = 0; o < H_; ++o) {
    float aq = 0.f, ak = 0.f, av = 0.f;
#pragma unroll
    for (int i = 0; i < H_; ++i) {
      aq += xc[i] * Wq[o * H_ + i];
      ak += xc[i] * Wk[o * H_ + i];
      av += xc[i] * Wv[o * H_ + i];
    }
    qb[(size_t)t * DPAD + o * 16 + c] = (_Float16)(aq * qscale);
    kb[(size_t)t * DPAD + o * 16 + c] = (_Float16)ak;
    vT[((size_t)b * DPAD + o * 16 + c) * N_ + n] = (_Float16)av;  // transposed
  }
  // zero the pad features 80..95 (c covers exactly 16 slots)
  qb[(size_t)t * DPAD + 80 + c] = (_Float16)0.f;
  kb[(size_t)t * DPAD + 80 + c] = (_Float16)0.f;
}

// -------------------------- flash attention (WMMA) --------------------------
// One wave = 32 query rows (2 A-tiles); 32 keys/iter:
//   12 WMMA (S, 2 qtiles x 2 key subtiles x 3 k-steps) + 10 WMMA (P*V).
// K/V fragments are loaded once per iteration and reused by both query tiles.
__global__ void __launch_bounds__(32)
flash_kernel(const _Float16* __restrict__ q,
             const _Float16* __restrict__ k,
             const _Float16* __restrict__ vT,
             float* __restrict__ oat) {
  __shared__ __align__(16) _Float16 sP[2][16][32];
  int wid = blockIdx.x;
  int b = wid / (N_ / 32);
  int qt = wid % (N_ / 32);
  int lane = threadIdx.x;
  int m = lane & 15;        // A-frag row / B-frag col
  int half = lane >> 4;     // which K half this lane carries
  int q0 = qt * 32;

  // Q A-fragments for both 16-row tiles: a[0..7]=feat half*8.., a[8..15]=+16
  v16h aq[2][3];
#pragma unroll
  for (int qa = 0; qa < 2; ++qa) {
    const _Float16* qbase = q + (size_t)(b * N_ + q0 + qa * 16 + m) * DPAD;
#pragma unroll
    for (int j = 0; j < 3; ++j) {
      half8 a0 = *(const half8*)(qbase + j * 32 + half * 8);
      half8 a1 = *(const half8*)(qbase + j * 32 + 16 + half * 8);
#pragma unroll
      for (int tt = 0; tt < 8; ++tt) { aq[qa][j][tt] = a0[tt]; aq[qa][j][8 + tt] = a1[tt]; }
    }
  }

  float mI[2][8], lI[2][8];
  v8f oacc[2][5];
#pragma unroll
  for (int qa = 0; qa < 2; ++qa) {
#pragma unroll
    for (int v = 0; v < 8; ++v) { mI[qa][v] = -1e30f; lI[qa][v] = 0.f; }
#pragma unroll
    for (int f = 0; f < 5; ++f) oacc[qa][f] = (v8f){};
  }

#pragma unroll 1
  for (int kb = 0; kb < N_ / 32; ++kb) {
    int kt0 = kb * 32;
    v8f s[2][2];  // [qtile][key subtile]

    // ---- S = Q K^T : load each K B-frag once, feed both query tiles ----
#pragma unroll
    for (int sub = 0; sub < 2; ++sub) {
      int kt = kt0 + sub * 16;
      v8f c0 = {}, c1 = {};
      const _Float16* kbase = k + (size_t)(b * N_ + kt + m) * DPAD;
#pragma unroll
      for (int j = 0; j < 3; ++j) {
        // B-frag: lane m = key col, b[0..7]=feat j*32+half*8.., b[8..15]=+16
        half8 b0 = *(const half8*)(kbase + j * 32 + half * 8);
        half8 b1 = *(const half8*)(kbase + j * 32 + 16 + half * 8);
        v16h bk;
#pragma unroll
        for (int tt = 0; tt < 8; ++tt) { bk[tt] = b0[tt]; bk[8 + tt] = b1[tt]; }
        c0 = __builtin_amdgcn_wmma_f32_16x16x32_f16(false, aq[0][j], false, bk,
                                                    (short)0, c0, false, false);
        c1 = __builtin_amdgcn_wmma_f32_16x16x32_f16(false, aq[1][j], false, bk,
                                                    (short)0, c1, false, false);
      }
      s[0][sub] = c0;
      s[1][sub] = c1;
    }

    // ---- streaming softmax (row = (v, half) across 16 lanes), per q-tile ----
    __syncthreads();
#pragma unroll
    for (int qa = 0; qa < 2; ++qa) {
      float alpha[8];
#pragma unroll
      for (int v = 0; v < 8; ++v) {
        float bm = fmaxf(s[qa][0][v], s[qa][1][v]);
#pragma unroll
        for (int mm = 1; mm < 16; mm <<= 1) bm = fmaxf(bm, __shfl_xor(bm, mm, 32));
        float mnew = fmaxf(mI[qa][v], bm);
        alpha[v] = __expf(mI[qa][v] - mnew);
        float p0 = __expf(s[qa][0][v] - mnew);
        float p1 = __expf(s[qa][1][v] - mnew);
        s[qa][0][v] = p0; s[qa][1][v] = p1;
        float rs = p0 + p1;
#pragma unroll
        for (int mm = 1; mm < 16; mm <<= 1) rs += __shfl_xor(rs, mm, 32);
        lI[qa][v] = lI[qa][v] * alpha[v] + rs;
        mI[qa][v] = mnew;
      }
#pragma unroll
      for (int f = 0; f < 5; ++f)
#pragma unroll
        for (int v = 0; v < 8; ++v) oacc[qa][f][v] *= alpha[v];

      // P: WMMA D layout -> LDS (A layout staging)
#pragma unroll
      for (int v = 0; v < 8; ++v) {
        int row = v + 8 * half;               // D layout: M = v + 8*(lane/16)
        sP[qa][row][m]      = (_Float16)s[qa][0][v];
        sP[qa][row][16 + m] = (_Float16)s[qa][1][v];
      }
    }
    __syncthreads();

    v16h ap[2];
#pragma unroll
    for (int qa = 0; qa < 2; ++qa) {
      half8 a0 = *(const half8*)&sP[qa][m][half * 8];
      half8 a1 = *(const half8*)&sP[qa][m][16 + half * 8];
#pragma unroll
      for (int tt = 0; tt < 8; ++tt) { ap[qa][tt] = a0[tt]; ap[qa][8 + tt] = a1[tt]; }
    }

    // ---- O += P * V : load each V B-frag once, feed both query tiles ----
#pragma unroll
    for (int f = 0; f < 5; ++f) {
      const _Float16* vbase =
          vT + ((size_t)b * DPAD + f * 16 + m) * N_ + kt0 + half * 8;
      half8 b0 = *(const half8*)(vbase);
      half8 b1 = *(const half8*)(vbase + 16);
      v16h bv;
#pragma unroll
      for (int tt = 0; tt < 8; ++tt) { bv[tt] = b0[tt]; bv[8 + tt] = b1[tt]; }
      oacc[0][f] = __builtin_amdgcn_wmma_f32_16x16x32_f16(false, ap[0], false, bv,
                                                          (short)0, oacc[0][f],
                                                          false, false);
      oacc[1][f] = __builtin_amdgcn_wmma_f32_16x16x32_f16(false, ap[1], false, bv,
                                                          (short)0, oacc[1][f],
                                                          false, false);
    }
  }

  // normalize and store O (f32, (B,N,80))
#pragma unroll
  for (int qa = 0; qa < 2; ++qa)
#pragma unroll
    for (int f = 0; f < 5; ++f)
#pragma unroll
      for (int v = 0; v < 8; ++v) {
        float val = oacc[qa][f][v] / lI[qa][v];
        oat[(size_t)(b * N_ + q0 + qa * 16 + v + 8 * half) * DFEAT + f * 16 + m] = val;
      }
}

// ------------------------ wo projection + residual --------------------------
__global__ void wo_res_kernel(const float* __restrict__ oin,
                              const float* __restrict__ wo,
                              float* __restrict__ h) {
  int tid = blockIdx.x * blockDim.x + threadIdx.x;
  int t = tid >> 4, c = tid & 15;
  if (t >= TOK) return;
  const float* orow = oin + (size_t)t * DFEAT;
  const float* W = wo + c_GRADE[c] * H_ * H_;
  float xi[H_];
#pragma unroll
  for (int i = 0; i < H_; ++i) xi[i] = orow[i * 16 + c];
#pragma unroll
  for (int o = 0; o < H_; ++o) {
    float a = 0.f;
#pragma unroll
    for (int i = 0; i < H_; ++i) a += xi[i] * W[o * H_ + i];
    h[(size_t)t * DFEAT + o * 16 + c] += a;
  }
}

// --------- layernorm + w1 + geometric product + gelu gate + w2 + res --------
__global__ void mlp_kernel(const float* __restrict__ hin,
                           const float* __restrict__ w1,
                           const float* __restrict__ w2,
                           float* __restrict__ h) {
  int tid = blockIdx.x * blockDim.x + threadIdx.x;
  int t = tid >> 4, c = tid & 15;
  if (t >= TOK) return;
  const float* hr = hin + (size_t)t * DFEAT;
  float inner = c_INNER[c];
  float xc[H_];
  float ip = 0.f;
#pragma unroll
  for (int i = 0; i < H_; ++i) { xc[i] = hr[i * 16 + c]; ip += xc[i] * xc[i] * inner; }
#pragma unroll
  for (int m = 1; m < 16; m <<= 1) ip += __shfl_xor(ip, m, 32);
  float inv = rsqrtf(ip * (1.0f / (H_ * 16.0f)) + 1e-6f);
#pragma unroll
  for (int i = 0; i < H_; ++i) xc[i] *= inv;

  int g = c_GRADE[c];
  const float* W1 = w1 + g * (2 * H_) * H_;
  float l[H_], r[H_];
#pragma unroll
  for (int o = 0; o < 2 * H_; ++o) {
    float a = 0.f;
#pragma unroll
    for (int i = 0; i < H_; ++i) a += xc[i] * W1[o * H_ + i];
    if (o < H_) l[o] = a; else r[o - H_] = a;
  }

  // geometric product: lane c computes blade component c; lanes cooperate
  int mk = c_MASK[c];
  float z[H_];
#pragma unroll
  for (int hh = 0; hh < H_; ++hh) {
    float lv = l[hh], rv = r[hh];
    float gp = 0.f;
#pragma unroll
    for (int i = 0; i < 16; ++i) {
      float li = __shfl(lv, i, 16);
      int mi = c_MASK[i];
      int mj = mi ^ mk;            // required r-blade so that mi^mj == mk
      int j = c_IDX[mj];
      float rj = __shfl(rv, j, 16);
      int sw = 0;
#pragma unroll
      for (int tb = 0; tb < 4; ++tb)
        if ((mj >> tb) & 1) sw += __popc(mi >> (tb + 1));
      float coef = (mi & mj & 1) ? 0.f : ((sw & 1) ? -1.f : 1.f);  // metric e0^2=0
      gp += coef * li * rj;
    }
    float g0 = __shfl(gp, 0, 16);  // scalar component
    float u = 0.7978845608028654f * (g0 + 0.044715f * g0 * g0 * g0);
    float gate = 0.5f * g0 * (1.0f + tanhf(u));  // gelu(tanh approx)
    z[hh] = gp * gate;
  }

  const float* W2 = w2 + g * H_ * H_;
#pragma unroll
  for (int o = 0; o < H_; ++o) {
    float a = 0.f;
#pragma unroll
    for (int i = 0; i < H_; ++i) a += z[i] * W2[o * H_ + i];
    h[(size_t)t * DFEAT + o * 16 + c] += a;
  }
}

// ------------------------- w_out + output extraction ------------------------
__global__ void out_kernel(const float* __restrict__ h,
                           const float* __restrict__ w_out,
                           float* __restrict__ out) {
  int t = blockIdx.x * blockDim.x + threadIdx.x;
  if (t >= TOK) return;
  const float* hr = h + (size_t)t * DFEAT;
  float* orow = out + (size_t)t * 35;
#pragma unroll 4
  for (int o = 0; o < 32; ++o) {
    float a = 0.f;
#pragma unroll
    for (int i = 0; i < H_; ++i) a += hr[i * 16 + 0] * w_out[(0 * 32 + o) * H_ + i];
    orow[o] = a;
  }
  const int tc[3] = {5, 6, 8};
#pragma unroll
  for (int d = 0; d < 3; ++d) {
    float a = 0.f;
#pragma unroll
    for (int i = 0; i < H_; ++i) a += hr[i * 16 + tc[d]] * w_out[(2 * 32 + 0) * H_ + i];
    orow[32 + d] = a;
  }
}

extern "C" void kernel_launch(void* const* d_in, const int* in_sizes, int n_in,
                              void* d_out, int out_size, void* d_ws, size_t ws_size,
                              hipStream_t stream) {
  const float* x     = (const float*)d_in[0];
  const float* w_in  = (const float*)d_in[1];
  const float* w_out = (const float*)d_in[2];
  const float* wq    = (const float*)d_in[3];
  const float* wk    = (const float*)d_in[4];
  const float* wv    = (const float*)d_in[5];
  const float* wo    = (const float*)d_in[6];
  const float* w1    = (const float*)d_in[7];
  const float* w2    = (const float*)d_in[8];

  char* ws = (char*)d_ws;
  float*     h   = (float*)ws;                       // 5,242,880 B
  float*     oat = (float*)(ws + 5242880);           // 5,242,880 B
  _Float16*  qb  = (_Float16*)(ws + 10485760);       // 3,145,728 B
  _Float16*  kb  = (_Float16*)(ws + 13631488);       // 3,145,728 B
  _Float16*  vT  = (_Float16*)(ws + 16777216);       // 3,145,728 B

  const int g16 = (TOK * 16) / 256;  // 1024 blocks of 256 (one thread per (t,c))
  embed_kernel<<<g16, 256, 0, stream>>>(x, w_in, h);
  for (int blk = 0; blk < 3; ++blk) {
    size_t w55 = (size_t)blk * 5 * H_ * H_;
    ln_qkv_kernel<<<g16, 256, 0, stream>>>(h, wq + w55, wk + w55, wv + w55,
                                           qb, kb, vT);
    flash_kernel<<<TOK / 32, 32, 0, stream>>>(qb, kb, vT, oat);
    wo_res_kernel<<<g16, 256, 0, stream>>>(oat, wo + w55, h);
    mlp_kernel<<<g16, 256, 0, stream>>>(h, w1 + (size_t)blk * 5 * 2 * H_ * H_,
                                        w2 + w55, h);
  }
  out_kernel<<<TOK / 256, 256, 0, stream>>>(h, w_out, (float*)d_out);
}
